// ToMe_84782654423140
// MI455X (gfx1250) — matrix operand order repaired and checked by reading, not compiled, not run on hardware
//
#include <hip/hip_runtime.h>
#include <hip/hip_bf16.h>

typedef _Float16 half8 __attribute__((ext_vector_type(8)));
typedef _Float16 v16h  __attribute__((ext_vector_type(16)));
typedef float    v8f   __attribute__((ext_vector_type(8)));

#define BSZ    32
#define NTOK   2048
#define CH     256
#define HALF_N 1024
#define RNUM   512
#define KEEP   1536
#define PITCH  264   // halves per LDS row: 256 + 8 pad (keeps 16B alignment, spreads banks)

// d_out float offsets (outputs concatenated flat in return order)
#define TGT_OFF  0
#define RP_OFF   12582912
#define AM_OFF   12779520
#define REFP_OFF 15138816
#define DN_OFF   15335424

// ---------------------------------------------------------------------------
// 1) L2-normalize each token row, convert to f16. One wave (32 lanes) / token.
// ---------------------------------------------------------------------------
__global__ __launch_bounds__(256) void k_normalize(const float* __restrict__ target,
                                                   _Float16* __restrict__ tn) {
  const int token = blockIdx.x * 8 + (threadIdx.x >> 5);
  const int lane  = threadIdx.x & 31;
  const float* row = target + (size_t)token * CH;
  float x[8]; float ss = 0.f;
#pragma unroll
  for (int i = 0; i < 8; ++i) { x[i] = row[lane + 32 * i]; ss += x[i] * x[i]; }
#pragma unroll
  for (int m = 16; m >= 1; m >>= 1) ss += __shfl_xor(ss, m, 32);
  const float inv = 1.0f / sqrtf(ss);
  _Float16* orow = tn + (size_t)token * CH;
#pragma unroll
  for (int i = 0; i < 8; ++i) orow[lane + 32 * i] = (_Float16)(x[i] * inv);
}

// ---------------------------------------------------------------------------
// 2) Batched scores = a . b^T with fused per-row max/argmax.
//    Block = 128 threads (4 waves). Block owns 64 A-rows; each wave a 16-row
//    M-tile. Iterate 64 N-tiles of 16 odd tokens; 8 chained WMMAs per tile.
//    Running max is kept PER LANE (each lane owns columns == mcol mod 16);
//    the 16-lane butterfly runs once at the end -> no ds_bpermute in the
//    inner loop, WMMAs issue back-to-back after one dscnt wait.
// ---------------------------------------------------------------------------
__global__ __launch_bounds__(128) void k_scores(const _Float16* __restrict__ tn,
                                                float* __restrict__ node_max,
                                                int* __restrict__ node_idx) {
  __shared__ _Float16 As[64 * PITCH];   // 33.0 KB
  __shared__ _Float16 Bs[16 * PITCH];   //  8.25 KB
  const int b      = blockIdx.x >> 4;          // 16 blocks per batch
  const int mblock = (blockIdx.x & 15) * 64;
  const int t      = threadIdx.x;
  const int wave   = t >> 5;
  const int lane   = t & 31;
  const _Float16* tb = tn + (size_t)b * NTOK * CH;

  // Stage 64 even-token rows into LDS (each thread: 16 contiguous f4 = 256B)
  {
    const int r  = t >> 1;
    const int hc = (t & 1) * 128;
    const float4* src = (const float4*)(tb + (size_t)(2 * (mblock + r)) * CH + hc);
    float4* dst = (float4*)(&As[r * PITCH + hc]);
#pragma unroll
    for (int q = 0; q < 16; ++q) dst[q] = src[q];
  }
  __syncthreads();

  const int mcol = lane & 15;   // A row / B column / D column within tile
  const int kh   = lane >> 4;   // half-wave selector per ISA fragment layout

  // Preload this wave's A fragments for all K (8 fragments of K=32) into VGPRs
  v16h afrag[8];
  {
    const _Float16* arow = &As[(wave * 16 + mcol) * PITCH];
#pragma unroll
    for (int kt = 0; kt < 8; ++kt) {
      const int k0 = kt * 32 + kh * 8;
      half8 lo = *(const half8*)(arow + k0);
      half8 hi = *(const half8*)(arow + k0 + 16);
      afrag[kt] = __builtin_shufflevector(lo, hi, 0,1,2,3,4,5,6,7,8,9,10,11,12,13,14,15);
    }
  }

  // Per-lane running (max, argmax) for rows v+8*kh over this lane's columns.
  float lmax[8]; int lidx[8];
#pragma unroll
  for (int v = 0; v < 8; ++v) { lmax[v] = -__builtin_inff(); lidx[v] = 0; }

  for (int nt = 0; nt < 64; ++nt) {
    __syncthreads();
    {  // stage 16 odd-token rows (B tile), 64B per thread
      const int r  = t >> 3;
      const int hc = (t & 7) * 32;
      const float4* src = (const float4*)(tb + (size_t)(2 * (nt * 16 + r) + 1) * CH + hc);
      float4* dst = (float4*)(&Bs[r * PITCH + hc]);
#pragma unroll
      for (int q = 0; q < 4; ++q) dst[q] = src[q];
    }
    __syncthreads();

    // Preload all 8 B fragments, then stream 8 WMMAs on the accumulator chain.
    v16h bfrag[8];
    {
      const _Float16* brow = &Bs[mcol * PITCH];
#pragma unroll
      for (int kt = 0; kt < 8; ++kt) {
        const int k0 = kt * 32 + kh * 8;
        half8 lo = *(const half8*)(brow + k0);
        half8 hi = *(const half8*)(brow + k0 + 16);
        bfrag[kt] = __builtin_shufflevector(lo, hi, 0,1,2,3,4,5,6,7,8,9,10,11,12,13,14,15);
      }
    }
    v8f c = {};
#pragma unroll
    for (int kt = 0; kt < 8; ++kt)
      c = __builtin_amdgcn_wmma_f32_16x16x32_f16(false, afrag[kt], false, bfrag[kt],
                                                 (short)0, c, false, false);

    // Branchless per-lane update. Columns this lane sees are strictly
    // increasing in nt, so strict '>' keeps the earliest (argmax-first).
    const int idx = nt * 16 + mcol;
#pragma unroll
    for (int v = 0; v < 8; ++v) {
      const bool g = c[v] > lmax[v];
      lmax[v] = g ? c[v] : lmax[v];
      lidx[v] = g ? idx  : lidx[v];
    }
  }

  // One 16-lane butterfly per row at the very end (val desc, idx asc on ties).
  const int rowbase = mblock + wave * 16 + 8 * kh;
#pragma unroll
  for (int v = 0; v < 8; ++v) {
    float val = lmax[v];
    int   idx = lidx[v];
#pragma unroll
    for (int m = 1; m <= 8; m <<= 1) {
      const float ov = __shfl_xor(val, m, 32);
      const int   oi = __shfl_xor(idx, m, 32);
      const bool take = (ov > val) || (ov == val && oi < idx);
      val = take ? ov : val;
      idx = take ? oi : idx;
    }
    if (mcol == 0) {
      node_max[b * HALF_N + rowbase + v] = val;
      node_idx[b * HALF_N + rowbase + v] = idx;
    }
  }
}

// ---------------------------------------------------------------------------
// 3) Per-batch bitonic sort (1024 keys) on composite key (-val, idx) ==
//    stable argsort(-node_max). Emits unm/src/dst indices; zeroes counts.
// ---------------------------------------------------------------------------
__global__ __launch_bounds__(1024) void k_sort(const float* __restrict__ node_max,
                                               const int* __restrict__ node_idx,
                                               int* __restrict__ unm_idx,
                                               int* __restrict__ src_idx,
                                               int* __restrict__ dst_idx,
                                               float* __restrict__ cnt) {
  __shared__ float sv[HALF_N];
  __shared__ int   si[HALF_N];
  const int b = blockIdx.x, t = threadIdx.x;
  sv[t] = node_max[b * HALF_N + t];
  si[t] = t;
  cnt[b * HALF_N + t] = 0.0f;
  __syncthreads();
  for (int k = 2; k <= HALF_N; k <<= 1) {
    for (int j = k >> 1; j > 0; j >>= 1) {
      const int ixj = t ^ j;
      if (ixj > t) {
        const float v1 = sv[t], v2 = sv[ixj];
        const int   i1 = si[t], i2 = si[ixj];
        const bool less2 = (v2 > v1) || (v2 == v1 && i2 < i1);  // elem@ixj before elem@t
        const bool ascending = ((t & k) == 0);
        if (less2 == ascending) { sv[t] = v2; si[t] = i2; sv[ixj] = v1; si[ixj] = i1; }
      }
      __syncthreads();
    }
  }
  const int e = si[t];
  if (t < RNUM) {
    src_idx[b * RNUM + t] = e;
    dst_idx[b * RNUM + t] = node_idx[b * HALF_N + e];
  } else {
    unm_idx[b * RNUM + (t - RNUM)] = e;
  }
}

// ---------------------------------------------------------------------------
// 4a) Initialize target_out: unm rows (gathered even tokens) + dst rows (odd).
// ---------------------------------------------------------------------------
__global__ __launch_bounds__(256) void k_init_out(const float* __restrict__ target,
                                                  const int* __restrict__ unm_idx,
                                                  float* __restrict__ out) {
  const int gw = blockIdx.x * 8 + (threadIdx.x >> 5);
  const int lane = threadIdx.x & 31;
  const int b = gw / KEEP, k = gw % KEEP;
  const int tok = (k < RNUM) ? (2 * unm_idx[b * RNUM + k]) : (2 * (k - RNUM) + 1);
  const float* src = target + ((size_t)b * NTOK + tok) * CH;
  float* dst = out + TGT_OFF + ((size_t)b * KEEP + k) * CH;
#pragma unroll
  for (int i = 0; i < 8; ++i) dst[lane + 32 * i] = src[lane + 32 * i];
}

// 4b) Scatter-add merged src rows + counts directly into d_out dst region.
__global__ __launch_bounds__(256) void k_scatter(const float* __restrict__ target,
                                                 const int* __restrict__ src_idx,
                                                 const int* __restrict__ dst_idx,
                                                 float* __restrict__ out,
                                                 float* __restrict__ cnt) {
  const int gw = blockIdx.x * 8 + (threadIdx.x >> 5);
  const int lane = threadIdx.x & 31;
  const int b = gw / RNUM, k = gw % RNUM;
  const int s = src_idx[b * RNUM + k];
  const int d = dst_idx[b * RNUM + k];
  const float* src = target + ((size_t)b * NTOK + 2 * s) * CH;
  float* dst = out + TGT_OFF + ((size_t)b * KEEP + RNUM + d) * CH;
#pragma unroll
  for (int i = 0; i < 8; ++i) atomicAdd(&dst[lane + 32 * i], src[lane + 32 * i]);
  if (lane == 0) atomicAdd(&cnt[b * HALF_N + d], 1.0f);
}

// 4c) dst_m = (dst + sum) / (1 + cnt)
__global__ __launch_bounds__(256) void k_scale(const float* __restrict__ cnt,
                                               float* __restrict__ out) {
  const int gw = blockIdx.x * 8 + (threadIdx.x >> 5);
  const int lane = threadIdx.x & 31;
  const int b = gw / HALF_N, j = gw % HALF_N;
  const float sc = 1.0f / (1.0f + cnt[b * HALF_N + j]);
  float* dst = out + TGT_OFF + ((size_t)b * KEEP + RNUM + j) * CH;
#pragma unroll
  for (int i = 0; i < 8; ++i) dst[lane + 32 * i] *= sc;
}

// ---------------------------------------------------------------------------
// 5) rp / refp / dn gathers (token map per batch; dn from batch 0 only)
// ---------------------------------------------------------------------------
__global__ __launch_bounds__(256) void k_small(const float* __restrict__ refpts,
                                               const float* __restrict__ ref_points,
                                               const int* __restrict__ dn_mask,
                                               const int* __restrict__ unm_idx,
                                               float* __restrict__ out) {
  const int gid = blockIdx.x * 256 + threadIdx.x;
  const int b = gid / KEEP, k = gid % KEEP;
  const int tok = (k < RNUM) ? (2 * unm_idx[b * RNUM + k]) : (2 * (k - RNUM) + 1);
  const float4 rv = *(const float4*)(refpts + ((size_t)b * NTOK + tok) * 4);
  *(float4*)(out + RP_OFF + ((size_t)b * KEEP + k) * 4) = rv;
  const float4 rv2 = *(const float4*)(ref_points + ((size_t)b * NTOK + tok) * 4);
  *(float4*)(out + REFP_OFF + ((size_t)b * KEEP + k) * 4) = rv2;
  if (b == 0) out[DN_OFF + k] = dn_mask[tok] ? 1.0f : 0.0f;
}

// 6) am[i][j] = atten_mask[map0(j)][map0(i)]  (the swapaxes transposes indexing)
__global__ __launch_bounds__(256) void k_am(const int* __restrict__ atten,
                                            const int* __restrict__ unm_idx,
                                            float* __restrict__ out) {
  const int gid = blockIdx.x * 256 + threadIdx.x;
  const int i = gid / KEEP, j = gid % KEEP;
  const int mi = (i < RNUM) ? (2 * unm_idx[i]) : (2 * (i - RNUM) + 1);
  const int mj = (j < RNUM) ? (2 * unm_idx[j]) : (2 * (j - RNUM) + 1);
  out[AM_OFF + (size_t)i * KEEP + j] = atten[(size_t)mj * NTOK + mi] ? 1.0f : 0.0f;
}

// ---------------------------------------------------------------------------
extern "C" void kernel_launch(void* const* d_in, const int* in_sizes, int n_in,
                              void* d_out, int out_size, void* d_ws, size_t ws_size,
                              hipStream_t stream) {
  const float* target     = (const float*)d_in[0];   // (32,2048,256) f32
  const float* refpts     = (const float*)d_in[1];   // (32,2048,1,4) f32
  const int*   atten      = (const int*)  d_in[2];   // (2048,2048) bool->int
  const float* ref_points = (const float*)d_in[3];   // (32,2048,4) f32
  const int*   dn_mask    = (const int*)  d_in[4];   // (2048,) bool->int
  float* out = (float*)d_out;
  char*  ws  = (char*)d_ws;

  _Float16* tn     = (_Float16*)ws;                 // 32*2048*256 f16 = 33,554,432 B
  float* node_max  = (float*)(ws + 33554432);       // 32*1024 f32
  int*   node_idx  = (int*)  (ws + 33685504);       // 32*1024 i32
  int*   unm_idx   = (int*)  (ws + 33816576);       // 32*512  i32
  int*   src_idx   = (int*)  (ws + 33882112);       // 32*512  i32
  int*   dst_idx   = (int*)  (ws + 33947648);       // 32*512  i32
  float* cnt       = (float*)(ws + 34013184);       // 32*1024 f32

  k_normalize<<<8192, 256, 0, stream>>>(target, tn);
  k_scores  <<< 512, 128, 0, stream>>>(tn, node_max, node_idx);
  k_sort    <<<  32, 1024, 0, stream>>>(node_max, node_idx, unm_idx, src_idx, dst_idx, cnt);
  k_init_out<<<6144, 256, 0, stream>>>(target, unm_idx, out);
  k_scatter <<<2048, 256, 0, stream>>>(target, src_idx, dst_idx, out, cnt);
  k_scale   <<<4096, 256, 0, stream>>>(cnt, out);
  k_small   <<< 192, 256, 0, stream>>>(refpts, ref_points, dn_mask, unm_idx, out);
  k_am      <<<9216, 256, 0, stream>>>(atten, unm_idx, out);
}
